// fSRN_71244917506711
// MI455X (gfx1250) — matrix operand order repaired and checked by reading, not compiled, not run on hardware
//
#include <hip/hip_runtime.h>

typedef _Float16 v16h __attribute__((ext_vector_type(16)));
typedef _Float16 v8h  __attribute__((ext_vector_type(8)));
typedef _Float16 v4h  __attribute__((ext_vector_type(4)));
typedef float    v8f  __attribute__((ext_vector_type(8)));
typedef unsigned int u32x4 __attribute__((ext_vector_type(4)));
typedef int          i32x4 __attribute__((ext_vector_type(4)));
typedef int          i32x8 __attribute__((ext_vector_type(8)));

union V16 { v16h v; v8h h[2]; };

static constexpr int   FD     = 128;   // feature dim
static constexpr int   TILE   = 128;   // rows (points) per workgroup
static constexpr int   AST    = 136;   // padded LDS row stride in halves (272B rows)
static constexpr int   NLAYER = 8;
static constexpr int   NMAT   = 16;    // 2 matrices per layer
static constexpr float OMEGA  = 30.0f;
static constexpr float INV2PI = 0.15915494309189535f;
static constexpr float WSCALE = OMEGA * INV2PI;   // fold OMEGA/(2pi) into weights

// ---------------------------------------------------------------------------
// TDM: DMA one 128x128 f16 matrix (contiguous 256B rows in global) into LDS
// with a 16B pad after every 256B row -> LDS stride 272B (= AST halves).
//   pad_interval = 5 (2^(5+1) = 64 DWORDs = 256B),  pad_amount = 3 (4 DWORDs)
// ---------------------------------------------------------------------------
__device__ __forceinline__ void tdm_load_w128(const _Float16* gsrc, unsigned lds_off) {
    unsigned long long ga = (unsigned long long)gsrc;
    u32x4 g0;
    g0[0] = 1u;                                           // count=1 valid descriptor
    g0[1] = lds_off;                                      // lds_addr (bytes)
    g0[2] = (unsigned)(ga & 0xFFFFFFFFu);                 // global_addr[31:0]
    g0[3] = (unsigned)((ga >> 32) & 0x01FFFFFFu) | (2u << 30); // addr[56:32] | type=2
    i32x8 g1;
    g1[0] = (1 << 16)      // data_size = 2 bytes
          | (1 << 20)      // pad_enable
          | (5 << 22)      // pad_interval: 64 DWORDs
          | (3 << 25);     // pad_amount:   4 DWORDs
    g1[1] = (FD << 16);    // tensor_dim0[15:0] in bits 63:48
    g1[2] = (FD << 16);    // tensor_dim0 hi = 0 ; tensor_dim1[15:0] in bits 95:80
    g1[3] = (FD << 16);    // tensor_dim1 hi = 0 ; tile_dim0 in bits 127:112
    g1[4] = FD;            // tile_dim1 = 128 ; tile_dim2 = 0
    g1[5] = FD;            // tensor_dim0_stride low 32 = 128 elements
    g1[6] = (FD * FD) << 16; // dim0_stride hi=0 ; tensor_dim1_stride low16
    g1[7] = 0;
    i32x4 z4 = {0, 0, 0, 0};
    i32x8 z8 = {0, 0, 0, 0, 0, 0, 0, 0};
    __builtin_amdgcn_tensor_load_to_lds(g0, g1, z4, z4, z8, 0);
}

// ---------------------------------------------------------------------------
// Pre-pass: convert W1/W2 (fp32) to f16 *pre-scaled by OMEGA/2pi*, slot s =
// (s even -> W1[s/2], s odd -> W2[s/2]).  65536 float4s total: <<<256,256>>>.
// ---------------------------------------------------------------------------
__global__ __launch_bounds__(256)
void convert_weights(const float* __restrict__ W1, const float* __restrict__ W2,
                     _Float16* __restrict__ dst) {
    int p = blockIdx.x * 256 + threadIdx.x;     // float4 index 0..65535
    int slot = p >> 12;                          // 4096 float4 per matrix
    int q = p & 4095;
    const float* base = (slot & 1) ? W2 : W1;
    const float4* src = (const float4*)(base + (slot >> 1) * FD * FD);
    float4 w = src[q];
    v4h hv = { (_Float16)(WSCALE * w.x), (_Float16)(WSCALE * w.y),
               (_Float16)(WSCALE * w.z), (_Float16)(WSCALE * w.w) };
    *(v4h*)(dst + (size_t)p * 4) = hv;
}

// ---------------------------------------------------------------------------
// Main fused kernel: TDM double-buffered weights, ping-pong activations,
// 2D wave tiling (each wave: 32 rows x 64 cols), WMMA f16->f32.
// ---------------------------------------------------------------------------
__global__ __launch_bounds__(256)
void siren_fused_tdm(const float* __restrict__ coords,
                     const float* __restrict__ W_first,
                     const float* __restrict__ b_first,
                     const _Float16* __restrict__ wf16,   // [16][128*128] f16 (pre-scaled)
                     const float* __restrict__ b1,
                     const float* __restrict__ b2,
                     const float* __restrict__ W_out,
                     const float* __restrict__ b_out,
                     float* __restrict__ out,
                     int N) {
    __shared__ __align__(16) _Float16 sW[2][FD * AST];     // weight ping-pong
    __shared__ __align__(16) _Float16 sAct[2][TILE * AST]; // activation ping-pong

    const int t    = threadIdx.x;
    const int l    = t & 31;
    const int wv   = t >> 5;
    const int ln   = l & 15;
    const int hi   = l >> 4;
    const int wm   = wv & 3;     // row block: 32 rows
    const int wn   = wv >> 2;    // col block: 64 cols
    const int row0 = blockIdx.x * TILE;

    // Kick off the DMA for matrix 0 immediately (overlaps first-layer work).
    if (wv == 0)
        tdm_load_w128(wf16, (unsigned)(unsigned long long)(void*)&sW[0][0]);

    // ---- Stage coords through sAct[1] (stage-0 acts land in sAct[0]) --------
    float* cf = (float*)&sAct[1][0];
    for (int e = t; e < TILE * 3; e += 256) {
        int r = e / 3, d = e - r * 3;
        int g = row0 + r;
        cf[e] = (g < N) ? coords[g * 3 + d] : 0.0f;
    }
    __syncthreads();

    float cx[2][8], cy[2][8], cz[2][8];
    #pragma unroll
    for (int mt = 0; mt < 2; ++mt)
        #pragma unroll
        for (int v = 0; v < 8; ++v) {
            int r = wm * 32 + mt * 16 + v + 8 * hi;   // D-layout row
            cx[mt][v] = cf[r * 3 + 0];
            cy[mt][v] = cf[r * 3 + 1];
            cz[mt][v] = cf[r * 3 + 2];
        }

    // ---- First SIREN layer (arg kept in revolutions for v_sin) --------------
    v8f hreg[2][4];
    #pragma unroll
    for (int mt = 0; mt < 2; ++mt)
        #pragma unroll
        for (int j = 0; j < 4; ++j) {
            int n = wn * 64 + j * 16 + ln;
            float w0 = WSCALE * W_first[n * 3 + 0];
            float w1 = WSCALE * W_first[n * 3 + 1];
            float w2 = WSCALE * W_first[n * 3 + 2];
            float bb = WSCALE * b_first[n];
            #pragma unroll
            for (int v = 0; v < 8; ++v) {
                float pre = fmaf(cx[mt][v], w0,
                            fmaf(cy[mt][v], w1, fmaf(cz[mt][v], w2, bb)));
                hreg[mt][j][v] = __builtin_amdgcn_sinf(pre);  // sin(2pi*pre)
            }
        }
    #pragma unroll
    for (int mt = 0; mt < 2; ++mt)
        #pragma unroll
        for (int j = 0; j < 4; ++j)
            #pragma unroll
            for (int v = 0; v < 8; ++v)
                sAct[0][(wm * 32 + mt * 16 + v + 8 * hi) * AST + wn * 64 + j * 16 + ln] =
                    (_Float16)hreg[mt][j][v];

    // ---- One pipeline stage: half a residual block --------------------------
    auto stage = [&](int st, const float* bias, bool second, float wout, bool writeNext) {
        if (wv == 0) __builtin_amdgcn_s_wait_tensorcnt(0);   // matrix st ready
        __syncthreads();
        if (wv == 0 && st + 1 < NMAT)                        // prefetch st+1
            tdm_load_w128(wf16 + (size_t)(st + 1) * FD * FD,
                          (unsigned)(unsigned long long)(void*)&sW[(st + 1) & 1][0]);
        const _Float16* wb   = &sW[st & 1][0];
        const _Float16* ain  = &sAct[st & 1][0];
        _Float16*       aout = &sAct[(st + 1) & 1][0];

        // Prefetch the 4 bias values (latency hides behind A-fragment loads).
        float ob[4];
        #pragma unroll
        for (int j = 0; j < 4; ++j)
            ob[j] = WSCALE * bias[wn * 64 + j * 16 + ln];

        // Hoist all A fragments for this wave's 32 rows into registers.
        V16 afr[2][4];
        #pragma unroll
        for (int mt = 0; mt < 2; ++mt)
            #pragma unroll
            for (int kc = 0; kc < 4; ++kc) {
                int base = (wm * 32 + mt * 16 + ln) * AST + kc * 32;
                afr[mt][kc].h[0] = *(const v8h*)&ain[base + hi * 8];
                afr[mt][kc].h[1] = *(const v8h*)&ain[base + 16 + hi * 8];
            }

        #pragma unroll
        for (int j = 0; j < 4; ++j) {
            const int n    = wn * 64 + j * 16 + ln;
            const int boff = n * AST;
            v8f acc0 = {}, acc1 = {};
            V16 bc, bn;
            bc.h[0] = *(const v8h*)&wb[boff + hi * 8];
            bc.h[1] = *(const v8h*)&wb[boff + 16 + hi * 8];
            #pragma unroll
            for (int kc = 0; kc < 4; ++kc) {
                if (kc < 3) {   // double-buffer B one kc ahead
                    bn.h[0] = *(const v8h*)&wb[boff + (kc + 1) * 32 + hi * 8];
                    bn.h[1] = *(const v8h*)&wb[boff + (kc + 1) * 32 + 16 + hi * 8];
                }
                acc0 = __builtin_amdgcn_wmma_f32_16x16x32_f16(
                           false, afr[0][kc].v, false, bc.v, (short)0, acc0, false, false);
                acc1 = __builtin_amdgcn_wmma_f32_16x16x32_f16(
                           false, afr[1][kc].v, false, bc.v, (short)0, acc1, false, false);
                bc = bn;
            }
            #pragma unroll
            for (int mt = 0; mt < 2; ++mt) {
                const v8f& acc = mt ? acc1 : acc0;
                v8f sv;
                #pragma unroll
                for (int v = 0; v < 8; ++v)               // 8 independent v_sin
                    sv[v] = __builtin_amdgcn_sinf(acc[v] + ob[j]);
                if (!second) {                            // s1 pass
                    #pragma unroll
                    for (int v = 0; v < 8; ++v)
                        aout[(wm * 32 + mt * 16 + v + 8 * hi) * AST + n] =
                            (_Float16)sv[v];
                } else {                                  // s2 pass + residual
                    #pragma unroll
                    for (int v = 0; v < 8; ++v)
                        hreg[mt][j][v] = wout * (hreg[mt][j][v] + sv[v]);
                    if (writeNext) {
                        #pragma unroll
                        for (int v = 0; v < 8; ++v)
                            aout[(wm * 32 + mt * 16 + v + 8 * hi) * AST + n] =
                                (_Float16)(0.5f * hreg[mt][j][v]);
                    }
                }
            }
        }
    };

    #pragma unroll 1
    for (int i = 0; i < NLAYER; ++i) {
        stage(2 * i,     b1 + i * FD, false, 0.0f, true);
        stage(2 * i + 1, b2 + i * FD, true,
              (i == NLAYER - 1) ? 0.5f : 1.0f, i < NLAYER - 1);
    }

    // ---- Output head: out = h @ W_out^T + b_out (O = 1) ---------------------
    float pr[2][8];
    #pragma unroll
    for (int mt = 0; mt < 2; ++mt)
        #pragma unroll
        for (int v = 0; v < 8; ++v) pr[mt][v] = 0.0f;
    #pragma unroll
    for (int j = 0; j < 4; ++j) {
        float wo = W_out[wn * 64 + j * 16 + ln];
        #pragma unroll
        for (int mt = 0; mt < 2; ++mt)
            #pragma unroll
            for (int v = 0; v < 8; ++v) pr[mt][v] += hreg[mt][j][v] * wo;
    }
    #pragma unroll
    for (int mt = 0; mt < 2; ++mt)
        #pragma unroll
        for (int v = 0; v < 8; ++v) {
            #pragma unroll
            for (int m = 8; m >= 1; m >>= 1)
                pr[mt][v] += __shfl_xor(pr[mt][v], m, 16);
        }
    float* po = (float*)&sAct[0][0];   // stage 15 read sAct[1]; sAct[0] is free
    if (ln == 0) {
        #pragma unroll
        for (int mt = 0; mt < 2; ++mt)
            #pragma unroll
            for (int v = 0; v < 8; ++v) {
                int r = wm * 32 + mt * 16 + v + 8 * hi;
                po[r * 2 + wn] = pr[mt][v];
            }
    }
    __syncthreads();
    if (t < TILE) {
        int g = row0 + t;
        if (g < N) out[g] = po[t * 2] + po[t * 2 + 1] + b_out[0];
    }
}

// ---------------------------------------------------------------------------
// Fallback (no workspace): synchronous f32->f16 weight staging, 1D tiling.
// Uses unscaled weights + __sinf.
// ---------------------------------------------------------------------------
__device__ __forceinline__ void load_weights_f16(const float* __restrict__ W,
                                                 _Float16* __restrict__ dst, int t) {
    const float4* src = (const float4*)W;
    #pragma unroll
    for (int i = 0; i < 16; ++i) {
        int p = t + i * 256;
        float4 w = src[p];
        int e = p << 2;
        int row = e >> 7, col = e & 127;
        v4h hv = { (_Float16)w.x, (_Float16)w.y, (_Float16)w.z, (_Float16)w.w };
        *(v4h*)&dst[row * AST + col] = hv;
    }
}

__global__ __launch_bounds__(256)
void siren_fused_sync(const float* __restrict__ coords,
                      const float* __restrict__ W_first,
                      const float* __restrict__ b_first,
                      const float* __restrict__ W1,
                      const float* __restrict__ b1,
                      const float* __restrict__ W2,
                      const float* __restrict__ b2,
                      const float* __restrict__ W_out,
                      const float* __restrict__ b_out,
                      float* __restrict__ out,
                      int N) {
    __shared__ __align__(16) _Float16 sW[FD * AST];
    __shared__ __align__(16) _Float16 sAct[TILE * AST];

    const int t    = threadIdx.x;
    const int l    = t & 31;
    const int wv   = t >> 5;
    const int ln   = l & 15;
    const int hi   = l >> 4;
    const int row0 = blockIdx.x * TILE;
    const int arow = wv * 16;

    float* cf = (float*)sW;
    for (int e = t; e < TILE * 3; e += 256) {
        int r = e / 3, d = e - r * 3;
        int g = row0 + r;
        cf[e] = (g < N) ? coords[g * 3 + d] : 0.0f;
    }
    __syncthreads();

    float cx[8], cy[8], cz[8];
    #pragma unroll
    for (int v = 0; v < 8; ++v) {
        int r = arow + v + 8 * hi;
        cx[v] = cf[r * 3 + 0];
        cy[v] = cf[r * 3 + 1];
        cz[v] = cf[r * 3 + 2];
    }

    v8f hreg[8];
    #pragma unroll
    for (int j = 0; j < 8; ++j) {
        int n = j * 16 + ln;
        float w0 = W_first[n * 3 + 0], w1 = W_first[n * 3 + 1], w2 = W_first[n * 3 + 2];
        float bb = b_first[n];
        #pragma unroll
        for (int v = 0; v < 8; ++v) {
            float pre = fmaf(cx[v], w0, fmaf(cy[v], w1, fmaf(cz[v], w2, bb)));
            hreg[j][v] = __sinf(OMEGA * pre);
        }
    }
    #pragma unroll
    for (int j = 0; j < 8; ++j)
        #pragma unroll
        for (int v = 0; v < 8; ++v)
            sAct[(arow + v + 8 * hi) * AST + j * 16 + ln] = (_Float16)hreg[j][v];

    const int aoff = (arow + ln) * AST;

    #pragma unroll 1
    for (int i = 0; i < NLAYER; ++i) {
        __syncthreads();
        load_weights_f16(W1 + i * FD * FD, sW, t);
        __syncthreads();

        v8f s1[8];
        #pragma unroll
        for (int j = 0; j < 8; ++j) {
            float obb = OMEGA * b1[i * FD + j * 16 + ln];
            v8f acc = {};
            const int boff = (j * 16 + ln) * AST;
            #pragma unroll
            for (int kc = 0; kc < 4; ++kc) {
                V16 a, b;
                a.h[0] = *(const v8h*)&sAct[aoff + kc * 32 + hi * 8];
                a.h[1] = *(const v8h*)&sAct[aoff + kc * 32 + 16 + hi * 8];
                b.h[0] = *(const v8h*)&sW[boff + kc * 32 + hi * 8];
                b.h[1] = *(const v8h*)&sW[boff + kc * 32 + 16 + hi * 8];
                acc = __builtin_amdgcn_wmma_f32_16x16x32_f16(
                          false, a.v, false, b.v, (short)0, acc, false, false);
            }
            #pragma unroll
            for (int v = 0; v < 8; ++v) s1[j][v] = __sinf(fmaf(OMEGA, acc[v], obb));
        }
        #pragma unroll
        for (int j = 0; j < 8; ++j)
            #pragma unroll
            for (int v = 0; v < 8; ++v)
                sAct[(arow + v + 8 * hi) * AST + j * 16 + ln] = (_Float16)s1[j][v];

        __syncthreads();
        load_weights_f16(W2 + i * FD * FD, sW, t);
        __syncthreads();

        const float wout = (i == NLAYER - 1) ? 0.5f : 1.0f;
        #pragma unroll
        for (int j = 0; j < 8; ++j) {
            float obb = OMEGA * b2[i * FD + j * 16 + ln];
            v8f acc = {};
            const int boff = (j * 16 + ln) * AST;
            #pragma unroll
            for (int kc = 0; kc < 4; ++kc) {
                V16 a, b;
                a.h[0] = *(const v8h*)&sAct[aoff + kc * 32 + hi * 8];
                a.h[1] = *(const v8h*)&sAct[aoff + kc * 32 + 16 + hi * 8];
                b.h[0] = *(const v8h*)&sW[boff + kc * 32 + hi * 8];
                b.h[1] = *(const v8h*)&sW[boff + kc * 32 + 16 + hi * 8];
                acc = __builtin_amdgcn_wmma_f32_16x16x32_f16(
                          false, a.v, false, b.v, (short)0, acc, false, false);
            }
            #pragma unroll
            for (int v = 0; v < 8; ++v) {
                float s2 = __sinf(fmaf(OMEGA, acc[v], obb));
                hreg[j][v] = wout * (hreg[j][v] + s2);
            }
        }
        if (i < NLAYER - 1) {
            #pragma unroll
            for (int j = 0; j < 8; ++j)
                #pragma unroll
                for (int v = 0; v < 8; ++v)
                    sAct[(arow + v + 8 * hi) * AST + j * 16 + ln] =
                        (_Float16)(0.5f * hreg[j][v]);
        }
    }

    float p[8];
    #pragma unroll
    for (int v = 0; v < 8; ++v) p[v] = 0.0f;
    #pragma unroll
    for (int j = 0; j < 8; ++j) {
        float wo = W_out[j * 16 + ln];
        #pragma unroll
        for (int v = 0; v < 8; ++v) p[v] += hreg[j][v] * wo;
    }
    #pragma unroll
    for (int v = 0; v < 8; ++v) {
        #pragma unroll
        for (int m = 8; m >= 1; m >>= 1)
            p[v] += __shfl_xor(p[v], m, 16);
    }
    if (ln == 0) {
        float bo = b_out[0];
        #pragma unroll
        for (int v = 0; v < 8; ++v) {
            int g = row0 + arow + v + 8 * hi;
            if (g < N) out[g] = p[v] + bo;
        }
    }
}

extern "C" void kernel_launch(void* const* d_in, const int* in_sizes, int n_in,
                              void* d_out, int out_size, void* d_ws, size_t ws_size,
                              hipStream_t stream) {
    (void)n_in; (void)out_size;
    const float* coords  = (const float*)d_in[0];
    const float* W_first = (const float*)d_in[1];
    const float* b_first = (const float*)d_in[2];
    const float* W1      = (const float*)d_in[3];
    const float* b1      = (const float*)d_in[4];
    const float* W2      = (const float*)d_in[5];
    const float* b2      = (const float*)d_in[6];
    const float* W_out   = (const float*)d_in[7];
    const float* b_out   = (const float*)d_in[8];
    float* out = (float*)d_out;

    const int N = in_sizes[0] / 3;
    const int blocks = (N + TILE - 1) / TILE;
    const size_t need = (size_t)NMAT * FD * FD * 2;  // 512KB of f16 weights

    if (ws_size >= need) {
        _Float16* wf16 = (_Float16*)d_ws;
        convert_weights<<<256, 256, 0, stream>>>(W1, W2, wf16);
        siren_fused_tdm<<<blocks, 256, 0, stream>>>(coords, W_first, b_first,
                                                    wf16, b1, b2, W_out, b_out,
                                                    out, N);
    } else {
        siren_fused_sync<<<blocks, 256, 0, stream>>>(coords, W_first, b_first,
                                                     W1, b1, W2, b2, W_out, b_out,
                                                     out, N);
    }
}
